// Zipformer2Encoder_15539191677137
// MI455X (gfx1250) — matrix operand order repaired and checked by reading, not compiled, not run on hardware
//
#include <hip/hip_runtime.h>
#include <hip/hip_bf16.h>

#define BATCH 16
#define HH 96
#define WWID 96
#define CC 512
#define BS 4
#define NBW 24
#define NB 576      // 24*24 blocks per image
#define HWP 9216    // 96*96 pixels per image
#define DD 64
#define TOPK 16
#define TK2 32
#define ALPHA_C 10.0f
#define EPS_C 1e-3f

typedef __bf16 bf16_t;
typedef __attribute__((ext_vector_type(16))) __bf16 v16bf;
typedef __attribute__((ext_vector_type(8)))  __bf16 v8bf;
typedef __attribute__((ext_vector_type(8)))  float  v8f;

static __device__ __forceinline__ unsigned rne_bf(float f) {
  unsigned u = __builtin_bit_cast(unsigned, f);
  return (u + 0x7FFFu + ((u >> 16) & 1u)) >> 16;       // round-to-nearest-even
}
static __device__ __forceinline__ bf16_t f2bf(float f) {
  unsigned short h = (unsigned short)rne_bf(f);
  return __builtin_bit_cast(bf16_t, h);
}
static __device__ __forceinline__ unsigned pack2bf(float lo, float hi) {
  return rne_bf(lo) | (rne_bf(hi) << 16);
}
static __device__ __forceinline__ v16bf join16(v8bf lo, v8bf hi) {
  return __builtin_shufflevector(lo, hi, 0, 1, 2, 3, 4, 5, 6, 7,
                                 8, 9, 10, 11, 12, 13, 14, 15);
}

// --------------------------------------------- one-shot weight fp32 -> bf16
__global__ void __launch_bounds__(256)
wprep_kernel(const float* __restrict__ W, unsigned* __restrict__ wbf32) {
  const int i = blockIdx.x * 256 + threadIdx.x;        // over DD*CC/2 pairs
  if (i < DD * CC / 2) {
    const float2 v = ((const float2*)W)[i];
    wbf32[i] = pack2bf(v.x, v.y);
  }
}

// ---------------------------------------------------------------- pooling
__global__ void __launch_bounds__(256)
pool_kernel(const float* __restrict__ src, float* __restrict__ bemb) {
  const int bg = blockIdx.x;                 // b*NB + blk
  const int b = bg / NB, blk = bg - b * NB;
  const int bh = blk / NBW, bw = blk - bh * NBW;
  const float* base = src + (((size_t)b * HH + bh * BS) * WWID + bw * BS) * CC;
  for (int c = threadIdx.x; c < CC; c += 256) {
    float s = 0.f;
#pragma unroll
    for (int i = 0; i < BS; ++i)
#pragma unroll
      for (int j = 0; j < BS; ++j)
        s += base[((size_t)i * WWID + j) * CC + c];
    bemb[(size_t)bg * CC + c] = s * (1.0f / 16.0f);
  }
}

// ---------------------- X(M x 512) @ Wbf(64 x 512)^T + bias -> bf16 (WMMA)
__global__ void __launch_bounds__(128)
proj_kernel(const float* __restrict__ X, const bf16_t* __restrict__ wbf,
            const float* __restrict__ bias, bf16_t* __restrict__ Y) {
  __shared__ bf16_t tile[16 * CC];           // 16 KB activation tile, bf16
  const int lane = threadIdx.x & 31;
  const int wave = threadIdx.x >> 5;         // N-tile 0..3
  const size_t row0 = (size_t)blockIdx.x * 16;

  // stage 16x512 fp32 rows as bf16 into LDS, packed 2-per-dword
  {
    const float2* xp2 = (const float2*)(X + row0 * CC);
    unsigned* t32 = (unsigned*)tile;
    for (int idx = threadIdx.x; idx < 16 * CC / 2; idx += 128) {
      const float2 v = xp2[idx];
      t32[idx] = pack2bf(v.x, v.y);
    }
  }
  __syncthreads();

  // A fragment (16-bit A 16x32): row = lane&15, K = e + (e<8?0:8) + (lane>=16?8:0)
  const int arow = lane & 15;
  const int ab = (lane >= 16) ? 8 : 0;
  // B fragment: N = lane&15, K = e + (e<8?0:8) + (lane>=16?8:0)
  const int n  = wave * 16 + (lane & 15);
  const int db = (lane >= 16) ? 16 : 0;
  const bf16_t* wrow = wbf + (size_t)n * CC;

  v8f acc = {};
#pragma unroll 4
  for (int c0 = 0; c0 < CC; c0 += 32) {
    const v8bf alo = *(const v8bf*)(&tile[arow * CC + c0 + ab]);        // ds b128
    const v8bf ahi = *(const v8bf*)(&tile[arow * CC + c0 + ab + 16]);   // ds b128
    const v16bf a  = join16(alo, ahi);
    const v16bf bm = join16(*(const v8bf*)(wrow + c0 + ab), *(const v8bf*)(wrow + c0 + ab + 16));
    acc = __builtin_amdgcn_wmma_f32_16x16x32_bf16(false, a, false, bm,
                                                  (short)0, acc, false, false);
  }
  const float bv = bias[n];
  const int mhi = (lane >= 16) ? 8 : 0;
#pragma unroll
  for (int r = 0; r < 8; ++r)
    Y[(row0 + r + mhi) * DD + n] = f2bf(acc[r] + bv);
}

// ---------------------------------- scores = q @ k^T, self-block masked (WMMA)
__global__ void __launch_bounds__(256)
scores_kernel(const bf16_t* __restrict__ qbf, const bf16_t* __restrict__ kbf,
              float* __restrict__ scores) {
  const int lane = threadIdx.x & 31;
  const int wave = threadIdx.x >> 5;          // 0..7
  const size_t R0 = (size_t)blockIdx.x * 16;  // global score row (b*NB + n)
  const int b  = (int)(R0 / NB);
  const int n0 = (int)(R0 - (size_t)b * NB);
  const bf16_t* kb = kbf + (size_t)b * HWP * DD;

  // A fragments (q rows, K=64 -> 2 chunks of 32), built once from 16B loads
  const int arow = lane & 15;
  const int ab = (lane >= 16) ? 8 : 0;
  const bf16_t* qrow = qbf + (R0 + arow) * DD;
  const v16bf a0 = join16(*(const v8bf*)(qrow + ab),
                          *(const v8bf*)(qrow + ab + 16));
  const v16bf a1 = join16(*(const v8bf*)(qrow + 32 + ab),
                          *(const v8bf*)(qrow + 32 + ab + 16));

  const int pn  = lane & 15;                  // N index = pixel within tile
  const int db  = (lane >= 16) ? 16 : 0;      // unused; B-fragment K base is ab
  const int mhi = (lane >= 16) ? 8 : 0;       // C-fragment row base

  for (int t = 0; t < (HWP / 16) / 8; ++t) {  // 72 pixel-tiles per wave
    const int p0 = (t * 8 + wave) * 16;
    const int p  = p0 + pn;
    // k row p is 64 contiguous bf16 (128 B): aligned 16-byte vector loads
    const v16bf bm0 = join16(*(const v8bf*)(kb + (size_t)p * DD + ab), *(const v8bf*)(kb + (size_t)p * DD + ab + 16));
    const v16bf bm1 = join16(*(const v8bf*)(kb + (size_t)p * DD + 32 + ab), *(const v8bf*)(kb + (size_t)p * DD + 32 + ab + 16));
    v8f c = {};
    c = __builtin_amdgcn_wmma_f32_16x16x32_bf16(false, a0, false, bm0, (short)0, c, false, false);
    c = __builtin_amdgcn_wmma_f32_16x16x32_bf16(false, a1, false, bm1, (short)0, c, false, false);

    const int ph = p / WWID, pw = p - ph * WWID;
    const int bid = (ph >> 2) * NBW + (pw >> 2);
#pragma unroll
    for (int r = 0; r < 8; ++r) {
      float v = c[r];
      if (bid == n0 + r + mhi) v = -__builtin_inff();   // mask own block
      // streaming output (340 MB, read exactly once): non-temporal
      __builtin_nontemporal_store(v, &scores[(R0 + r + mhi) * (size_t)HWP + p]);
    }
  }
}

// ------------------------------------------------ top-32 per score row
__global__ void __launch_bounds__(256)
topk_kernel(const float* __restrict__ scores, float* __restrict__ svals,
            int* __restrict__ sidx) {
  __shared__ float sv[HWP];                   // 36 KB (of 320 KB LDS)
  __shared__ float rv[256];
  __shared__ int   ri[256];
  const int row = blockIdx.x;
  const int t = threadIdx.x;
  const float* srow = scores + (size_t)row * HWP;
  for (int p = t; p < HWP; p += 256)
    sv[p] = __builtin_nontemporal_load(&srow[p]);
  __syncthreads();
  for (int r = 0; r < TK2; ++r) {
    float bv = -__builtin_inff(); int bi = 0x7FFFFFFF;
    for (int p = t; p < HWP; p += 256) {
      float v = sv[p];
      if (v > bv || (v == bv && p < bi)) { bv = v; bi = p; }
    }
    rv[t] = bv; ri[t] = bi;
    __syncthreads();
    for (int s = 128; s > 0; s >>= 1) {
      if (t < s) {
        float ov = rv[t + s]; int oi = ri[t + s];
        if (ov > rv[t] || (ov == rv[t] && oi < ri[t])) { rv[t] = ov; ri[t] = oi; }
      }
      __syncthreads();
    }
    if (t == 0) {
      svals[(size_t)row * TK2 + r] = rv[0];
      sidx [(size_t)row * TK2 + r] = ri[0];
      sv[ri[0]] = -__builtin_inff();
    }
    __syncthreads();
  }
}

// ------------------------------------------------ per-batch diff RMS
__global__ void __launch_bounds__(256)
rms_kernel(const float* __restrict__ svals, float* __restrict__ rms) {
  __shared__ float red[256];
  const int b = blockIdx.x, t = threadIdx.x;
  const float* sb = svals + (size_t)b * NB * TK2;
  float sum = 0.f;
  for (int idx = t; idx < NB * TOPK * TK2; idx += 256) {
    const int n = idx / (TOPK * TK2);
    const int rem = idx - n * (TOPK * TK2);
    const int i = rem / TK2;
    const int j = rem - i * TK2;
    const float d = sb[n * TK2 + i] - sb[n * TK2 + j];
    sum += d * d;
  }
  red[t] = sum;
  __syncthreads();
  for (int s = 128; s > 0; s >>= 1) {
    if (t < s) red[t] += red[t + s];
    __syncthreads();
  }
  if (t == 0) rms[b] = __builtin_sqrtf(red[0] / (float)(NB * TOPK * TK2));
}

// ------------------------------------------------ weights + index output
__global__ void __launch_bounds__(256)
weights_kernel(const float* __restrict__ svals, const int* __restrict__ sidx,
               const float* __restrict__ rms, float* __restrict__ out) {
  const int idx = blockIdx.x * 256 + threadIdx.x;   // over B*NB*TOPK
  if (idx >= BATCH * NB * TOPK) return;
  const int row = idx / TOPK;
  const int i = idx - row * TOPK;
  const int b = row / NB;
  const float scale = ALPHA_C / (rms[b] + EPS_C);
  const float si = svals[(size_t)row * TK2 + i];
  float s = 0.f;
#pragma unroll
  for (int j = 0; j < TK2; ++j) {
    const float d = (si - svals[(size_t)row * TK2 + j]) * scale;
    s += 1.0f / (1.0f + __expf(-d));
  }
  out[idx] = (float)sidx[(size_t)row * TK2 + i];              // indexes first
  out[BATCH * NB * TOPK + idx] = tanhf(s - (float)TOPK);      // then weights
}

extern "C" void kernel_launch(void* const* d_in, const int* in_sizes, int n_in,
                              void* d_out, int out_size, void* d_ws, size_t ws_size,
                              hipStream_t stream) {
  const float* src = (const float*)d_in[0];
  const float* Wq  = (const float*)d_in[1];
  const float* bq  = (const float*)d_in[2];
  const float* Wk  = (const float*)d_in[3];
  const float* bk  = (const float*)d_in[4];

  char* ws = (char*)d_ws;
  float*  bemb  = (float*)ws;   ws += (size_t)BATCH * NB  * CC  * 4;   // 18.9 MB
  bf16_t* qbf   = (bf16_t*)ws;  ws += (size_t)BATCH * NB  * DD  * 2;   // 1.2 MB
  bf16_t* kbf   = (bf16_t*)ws;  ws += (size_t)BATCH * HWP * DD  * 2;   // 18.9 MB
  bf16_t* wqbf  = (bf16_t*)ws;  ws += (size_t)DD * CC * 2;             // 64 KB
  bf16_t* wkbf  = (bf16_t*)ws;  ws += (size_t)DD * CC * 2;             // 64 KB
  float*  svals = (float*)ws;   ws += (size_t)BATCH * NB  * TK2 * 4;   // 1.2 MB
  int*    sidx  = (int*)ws;     ws += (size_t)BATCH * NB  * TK2 * 4;   // 1.2 MB
  float*  rms   = (float*)ws;   ws += 64;
  float*  scores = (float*)ws;                                         // 340 MB

  wprep_kernel  <<<(DD * CC / 2 + 255) / 256, 256, 0, stream>>>(Wq, (unsigned*)wqbf);
  wprep_kernel  <<<(DD * CC / 2 + 255) / 256, 256, 0, stream>>>(Wk, (unsigned*)wkbf);
  pool_kernel   <<<BATCH * NB,         256, 0, stream>>>(src, bemb);
  proj_kernel   <<<(BATCH * HWP) / 16, 128, 0, stream>>>(src,  wkbf, bk, kbf);
  proj_kernel   <<<(BATCH * NB)  / 16, 128, 0, stream>>>(bemb, wqbf, bq, qbf);
  scores_kernel <<<(BATCH * NB)  / 16, 256, 0, stream>>>(qbf, kbf, scores);
  topk_kernel   <<<BATCH * NB,         256, 0, stream>>>(scores, svals, sidx);
  rms_kernel    <<<BATCH,              256, 0, stream>>>(svals, rms);
  weights_kernel<<<(BATCH * NB * TOPK + 255) / 256, 256, 0, stream>>>(svals, sidx, rms, (float*)d_out);
}
